// GaussianMasks_5634997093107
// MI455X (gfx1250) — matrix operand order repaired and checked by reading, not compiled, not run on hardware
//
#include <hip/hip_runtime.h>

typedef _Float16 f16;
typedef __attribute__((ext_vector_type(16))) _Float16 v16h;
typedef __attribute__((ext_vector_type(8)))  _Float16 v8h;
typedef __attribute__((ext_vector_type(8)))  float    v8f;

#define AP_LD 136   // f16 stride for prob matrix rows (bank-spread padding)
#define BT_LD 136   // f16 stride for transposed V rows
#define SMEM_BYTES (128*AP_LD*2 + 64*BT_LD*2 + 512*4)

// One block = one batched-GEMM slice: out[128x64] = softmax(attn - gauss)[128x128] * V[128x64]
// MODE 0: x-part, slice (b, r): V rows contiguous (stride 64), out written.
// MODE 1: y-part, slice (b, c): V rows strided (stride 8192), out accumulated.
template <int MODE>
__global__ __launch_bounds__(256) void axial_gauss_attn_kernel(
    const float* __restrict__ attn,    // [1024][128][128], slice-contiguous for both passes
    const float* __restrict__ value,   // [8][128][128][64]
    float* __restrict__ out,           // [8][128][128][64]
    const float* __restrict__ shiftp,
    const float* __restrict__ biasp)
{
    extern __shared__ char smem[];
    f16*   aP  = (f16*)smem;                               // [128][AP_LD] f16 probs
    f16*   bT  = (f16*)(smem + 128*AP_LD*2);               // [64][BT_LD]  f16 V^T
    float* red = (float*)(smem + 128*AP_LD*2 + 64*BT_LD*2);// [512] reduction scratch

    const int tid = threadIdx.x;
    const int s   = blockIdx.x;
    const int b   = s >> 7;
    const int rc  = s & 127;

    const float shift = shiftp[0];
    const float bias  = biasp[0];

    const float* aslice = attn + (size_t)s * (128 * 128);

    // ---------------- softmax (fp32, axis -1) with gaussian penalty ----------------
    {
        const int   row  = tid >> 1;
        const int   c0   = (tid & 1) * 64;
        const float frow = (float)row;
        const float* ar  = aslice + (size_t)row * 128 + c0;

        float m = -3.0e38f;
        #pragma unroll 8
        for (int j = 0; j < 64; ++j) {
            float dj = (float)(c0 + j) - frow;
            float l  = ar[j] - (shift * dj * dj + bias);
            m = fmaxf(m, l);
        }
        red[tid] = m;
        __syncthreads();
        m = fmaxf(red[row * 2], red[row * 2 + 1]);
        __syncthreads();

        float sum = 0.0f;
        f16* prow = aP + row * AP_LD + c0;
        #pragma unroll 8
        for (int j = 0; j < 64; ++j) {
            float dj = (float)(c0 + j) - frow;
            float l  = ar[j] - (shift * dj * dj + bias);
            float e  = __expf(l - m);
            sum += e;
            prow[j] = (f16)e;
        }
        red[256 + tid] = sum;
        __syncthreads();
        float inv = 1.0f / (red[256 + row * 2] + red[256 + row * 2 + 1]);
        #pragma unroll 8
        for (int j = 0; j < 64; ++j)
            prow[j] = (f16)((float)prow[j] * inv);
    }

    // ---------------- load V slice -> f16, transposed (bT[d][k]) ----------------
    constexpr size_t kSliceStride = 128 * 64;        // floats per (b, r/c) slice
    constexpr int    kVStride     = (MODE == 0) ? 64 : 128 * 64;
    const size_t vOff = (MODE == 0) ? (size_t)s * kSliceStride
                                    : (size_t)b * (128 * kSliceStride) + (size_t)rc * 64;
    const float* vsrc = value + vOff;
    for (int i = tid; i < 128 * 64; i += 256) {
        int k = i >> 6;   // K index 0..127
        int d = i & 63;   // N index 0..63
        bT[d * BT_LD + k] = (f16)vsrc[(size_t)k * kVStride + d];
    }
    __syncthreads();

    // ---------------- WMMA: D[128x64] = A[128x128] * B[128x64] (fp32 acc) ----------------
    const int wave = tid >> 5;   // M tile 0..7
    const int lane = tid & 31;
    const int lm   = lane & 15;
    const int hi   = lane >> 4;

    // A 16x32 f16 layout: lane half 'hi' owns K = hi*8..hi*8+7 (VGPR0-3) and 16+hi*8.. (VGPR4-7)
    v16h afrag[4];
    #pragma unroll
    for (int kb = 0; kb < 4; ++kb) {
        const f16* pA = aP + (wave * 16 + lm) * AP_LD + kb * 32 + hi * 8;
        v8h lo = *(const v8h*)(pA);
        v8h hh = *(const v8h*)(pA + 16);
        afrag[kb] = __builtin_shufflevector(lo, hh, 0,1,2,3,4,5,6,7,8,9,10,11,12,13,14,15);
    }

    constexpr int kOStride = (MODE == 0) ? 64 : 128 * 64;
    const size_t oOff = (MODE == 0) ? (size_t)s * kSliceStride
                                    : (size_t)b * (128 * kSliceStride) + (size_t)rc * 64;
    float* optr = out + oOff;

    #pragma unroll
    for (int nt = 0; nt < 4; ++nt) {
        v8f acc = {};
        #pragma unroll
        for (int kb = 0; kb < 4; ++kb) {
            // B 32x16 f16 layout: lane half 'hi' owns K = hi*8..hi*8+7 and 16+hi*8..16+hi*8+7 (matches A)
            const f16* pB = bT + (nt * 16 + lm) * BT_LD + kb * 32 + hi * 8;
            v8h b0 = *(const v8h*)(pB);
            v8h b1 = *(const v8h*)(pB + 16);
            v16h bfrag = __builtin_shufflevector(b0, b1, 0,1,2,3,4,5,6,7,8,9,10,11,12,13,14,15);
            acc = __builtin_amdgcn_wmma_f32_16x16x32_f16(
                /*neg_a=*/false, afrag[kb], /*neg_b=*/false, bfrag,
                /*c_mod=*/(short)0, acc, /*reuse_a=*/false, /*reuse_b=*/false);
        }
        const int rbase = wave * 16 + hi * 8;   // C/D layout: VGPR i -> M = 8*hi + i
        const int col   = nt * 16 + lm;
        if constexpr (MODE == 0) {
            #pragma unroll
            for (int i = 0; i < 8; ++i)
                optr[(size_t)(rbase + i) * kOStride + col] = acc[i];
        } else {
            // Batch the read-modify-write: issue all 8 loads, single wait, then add+store.
            float old[8];
            #pragma unroll
            for (int i = 0; i < 8; ++i)
                old[i] = optr[(size_t)(rbase + i) * kOStride + col];
            #pragma unroll
            for (int i = 0; i < 8; ++i)
                optr[(size_t)(rbase + i) * kOStride + col] = old[i] + acc[i];
        }
    }
}

extern "C" void kernel_launch(void* const* d_in, const int* in_sizes, int n_in,
                              void* d_out, int out_size, void* d_ws, size_t ws_size,
                              hipStream_t stream) {
    (void)in_sizes; (void)n_in; (void)d_ws; (void)ws_size; (void)out_size;
    // setup_inputs order: x, atten_x_full, atten_y_full, value_full, shift, bias
    const float* attn_x = (const float*)d_in[1];
    const float* attn_y = (const float*)d_in[2];
    const float* value  = (const float*)d_in[3];
    const float* shiftp = (const float*)d_in[4];
    const float* biasp  = (const float*)d_in[5];
    float* out = (float*)d_out;

    dim3 grid(1024), block(256);
    // Pass 0: x-part writes out; Pass 1: y-part accumulates (stream-ordered, no races).
    axial_gauss_attn_kernel<0><<<grid, block, SMEM_BYTES, stream>>>(attn_x, value, out, shiftp, biasp);
    axial_gauss_attn_kernel<1><<<grid, block, SMEM_BYTES, stream>>>(attn_y, value, out, shiftp, biasp);
}
